// LightweightAttention_69784628626053
// MI455X (gfx1250) — matrix operand order repaired and checked by reading, not compiled, not run on hardware
//
#include <hip/hip_runtime.h>
#include <hip/hip_bf16.h>
#include <stddef.h>
#include <stdint.h>

// ---------------------------------------------------------------------------
// Types / helpers for CDNA5 (gfx1250) WMMA, wave32
// ---------------------------------------------------------------------------
typedef __attribute__((ext_vector_type(16))) __bf16 v16bf;
typedef __attribute__((ext_vector_type(8)))  __bf16 v8bf;
typedef __attribute__((ext_vector_type(4)))  __bf16 v4bf;
typedef __attribute__((ext_vector_type(8)))  float  v8f;

__device__ __forceinline__ __bf16 f2bf(float f) {
    unsigned u = __builtin_bit_cast(unsigned, f);
    u += 0x7fffu + ((u >> 16) & 1u);              // round-to-nearest-even
    unsigned short h = (unsigned short)(u >> 16);
    return __builtin_bit_cast(__bf16, h);
}

// D = A(16x32 bf16) x B(32x16 bf16) + C, fp32 accumulate
__device__ __forceinline__ v8f wmma_bf16(v16bf a, v16bf b, v8f c) {
    return __builtin_amdgcn_wmma_f32_16x16x32_bf16(
        false, a, false, b, (short)0, c, false, false);
}

__device__ __forceinline__ v16bf cat16(v8bf lo, v8bf hi) {
    v16bf r;
#pragma unroll
    for (int i = 0; i < 8; ++i) { r[i] = lo[i]; r[8 + i] = hi[i]; }
    return r;
}

// A operand: 16x32 tile, element (m,k) at src[m*ld + k] (row-major).
__device__ __forceinline__ v16bf load_a(const __bf16* src, int ld, int lane) {
    int row = lane & 15;
    int kb  = (lane >> 4) << 3;        // 0 or 8
    const __bf16* p = src + (size_t)row * ld + kb;
    return cat16(*(const v8bf*)p, *(const v8bf*)(p + 16));
}

// B operand: 32x16 tile, element (k,n) at src[n*ld + k] (B^T row-major).
__device__ __forceinline__ v16bf load_b(const __bf16* src, int ld, int lane) {
    int col = lane & 15;
    int kb  = (lane >> 4) << 3;        // 0 or 8
    const __bf16* p = src + (size_t)col * ld + kb;
    return cat16(*(const v8bf*)p, *(const v8bf*)(p + 16));
}

// ---- CDNA5 async global -> LDS copy (ASYNCcnt path) -----------------------
__device__ __forceinline__ void async_ld_b128(unsigned lds_off, const void* g) {
    asm volatile("global_load_async_to_lds_b128 %0, %1, off"
                 :: "v"(lds_off), "v"((unsigned long long)(uintptr_t)g)
                 : "memory");
}
__device__ __forceinline__ void wait_async0() {
    asm volatile("s_wait_asynccnt 0" ::: "memory");
}
__device__ __forceinline__ unsigned lds_off32(const void* p) {
    return (unsigned)(uintptr_t)p;     // flat shared addr low 32 bits = LDS offset
}

// ---------------------------------------------------------------------------
// Constants
// ---------------------------------------------------------------------------
#define BB   16
#define CC   512
#define NN   1024           // H*W
#define HH   8              // heads
#define DD   64             // head dim
#define O3   (3 * CC)       // 1536

// ---------------------------------------------------------------------------
// Kernel 0: fp32 -> bf16 weight downconvert
// ---------------------------------------------------------------------------
__global__ void la_cvt_bf16(const float* __restrict__ in, __bf16* __restrict__ out, int n) {
    int i = blockIdx.x * blockDim.x + threadIdx.x;
    if (i < n) out[i] = f2bf(in[i]);
}

// ---------------------------------------------------------------------------
// Kernel 1: GroupNorm (groups = 8), output bf16 token-major [B,N,C]
// ---------------------------------------------------------------------------
__global__ void la_groupnorm(const float* __restrict__ x,
                             const float* __restrict__ gw,
                             const float* __restrict__ gb,
                             __bf16* __restrict__ xn) {   // [B,N,C]
    const int chPerG = CC / HH;                    // 64
    const int tot    = chPerG * NN;                // 65536
    int bg = blockIdx.x;
    int g  = bg & 7;
    int b  = bg >> 3;
    const float* xp = x + (size_t)b * CC * NN + (size_t)g * chPerG * NN;  // [64,1024]
    __bf16*      op = xn + (size_t)b * NN * CC + (size_t)g * chPerG;      // +n*512

    float s = 0.f, s2 = 0.f;
    for (int i = threadIdx.x; i < tot / 4; i += blockDim.x) {
        float4 v = ((const float4*)xp)[i];
        s  += v.x + v.y + v.z + v.w;
        s2 += v.x * v.x + v.y * v.y + v.z * v.z + v.w * v.w;
    }
    __shared__ float rs[256], rq[256];
    rs[threadIdx.x] = s; rq[threadIdx.x] = s2;
    __syncthreads();
    for (int off = 128; off > 0; off >>= 1) {
        if (threadIdx.x < off) {
            rs[threadIdx.x] += rs[threadIdx.x + off];
            rq[threadIdx.x] += rq[threadIdx.x + off];
        }
        __syncthreads();
    }
    float mu  = rs[0] * (1.f / tot);
    float var = rq[0] * (1.f / tot) - mu * mu;
    float inv = rsqrtf(var + 1e-5f);

    __shared__ __bf16 tile[64][68];                // 64c x 64n, padded
    for (int t = 0; t < 16; ++t) {
        int n0 = t * 64;
#pragma unroll
        for (int r = 0; r < 4; ++r) {
            int lin = r * 256 + threadIdx.x;       // 0..1023
            int c   = lin >> 4;                    // 0..63
            int jj  = (lin & 15) * 4;              // 0..60
            float4 v = *(const float4*)(xp + (size_t)c * NN + n0 + jj);
            float wc = gw[g * chPerG + c];
            float bc = gb[g * chPerG + c];
            tile[c][jj + 0] = f2bf((v.x - mu) * inv * wc + bc);
            tile[c][jj + 1] = f2bf((v.y - mu) * inv * wc + bc);
            tile[c][jj + 2] = f2bf((v.z - mu) * inv * wc + bc);
            tile[c][jj + 3] = f2bf((v.w - mu) * inv * wc + bc);
        }
        __syncthreads();
#pragma unroll
        for (int r = 0; r < 4; ++r) {
            int lin = r * 256 + threadIdx.x;
            int j   = lin >> 4;
            int cq  = (lin & 15) * 4;
            v4bf o;
            o[0] = tile[cq + 0][j];
            o[1] = tile[cq + 1][j];
            o[2] = tile[cq + 2][j];
            o[3] = tile[cq + 3][j];
            *(v4bf*)(op + (size_t)(n0 + j) * CC + cq) = o;
        }
        __syncthreads();
    }
}

// ---------------------------------------------------------------------------
// Kernel 2: QKV GEMM, double-buffered async B staging.
// Block = 8 waves, 8 consecutive M-tiles sharing one (b, 64-token strip).
// Pipeline: prologue stages chunk0; loop = wait / barrier / issue-next /
// compute, so the async DMA overlaps the 4 WMMAs of the current chunk.
// blocks = 12(mg) * 16(nt4) * 16(b) = 3072 x 256 threads
// ---------------------------------------------------------------------------
__global__ void la_qkv(const __bf16* __restrict__ wq,     // [1536,512] bf16
                       const float*  __restrict__ qbias,  // [1536]
                       const __bf16* __restrict__ xn,     // [B,N,C] token-major
                       __bf16* __restrict__ q,
                       __bf16* __restrict__ k,
                       __bf16* __restrict__ v) {
    __shared__ __bf16 bstage[2][64 * 32];          // [n][c] 4KB x2
    int blk = blockIdx.x;
    int mg  = blk % 12;
    int rem = blk / 12;
    int nt4 = rem & 15;
    int b   = rem >> 4;
    int wid  = threadIdx.x >> 5;
    int lane = threadIdx.x & 31;
    int mt   = mg * 8 + wid;                       // 0..95

    const __bf16* A  = wq + (size_t)mt * 16 * CC;
    const __bf16* Bb = xn + (size_t)b * NN * CC + (size_t)nt4 * 64 * CC;

    int t  = threadIdx.x;
    int sn = t >> 2;                               // 0..63 token row
    int sp = t & 3;                                // 16B chunk in row
    const __bf16* gsrc = Bb + (size_t)sn * CC + sp * 8;

    // prologue: stage chunk 0 into buffer 0
    async_ld_b128(lds_off32(bstage[0]) + (unsigned)t * 16, gsrc);

    v8f acc0 = {}, acc1 = {}, acc2 = {}, acc3 = {};
    for (int k0 = 0, it = 0; k0 < CC; k0 += 32, ++it) {
        wait_async0();                             // previous chunk landed
        __syncthreads();                           // all waves ready; old buf free
        if (k0 + 32 < CC)                          // overlap next DMA with WMMAs
            async_ld_b128(lds_off32(bstage[(it + 1) & 1]) + (unsigned)t * 16,
                          gsrc + k0 + 32);

        const __bf16* bs = bstage[it & 1];
        v16bf a = load_a(A + k0, CC, lane);
        if (k0 + 32 < CC)
            __builtin_prefetch(A + (size_t)(lane & 15) * CC + k0 + 32, 0, 1);
        acc0 = wmma_bf16(a, load_b(bs + 0 * 16 * 32, 32, lane), acc0);
        acc1 = wmma_bf16(a, load_b(bs + 1 * 16 * 32, 32, lane), acc1);
        acc2 = wmma_bf16(a, load_b(bs + 2 * 16 * 32, 32, lane), acc2);
        acc3 = wmma_bf16(a, load_b(bs + 3 * 16 * 32, 32, lane), acc3);
    }

    int col = lane & 15;
    int hi  = lane >> 4;
    v8f* accs[4] = { &acc0, &acc1, &acc2, &acc3 };
#pragma unroll
    for (int j = 0; j < 4; ++j) {
        int n = nt4 * 64 + j * 16 + col;
#pragma unroll
        for (int r = 0; r < 8; ++r) {
            int   o   = mt * 16 + r + 8 * hi;
            float val = (*accs[j])[r] + qbias[o];
            int   c   = o & (CC - 1);
            int   wch = o >> 9;                    // 0=q,1=k,2=v
            int   h   = c >> 6;
            int   ci  = c & 63;
            size_t bh = (size_t)(b * HH + h);
            if (wch == 0) {
                q[(bh * NN + n) * DD + ci] = f2bf(val * 0.125f);
            } else if (wch == 1) {
                k[(bh * NN + n) * DD + ci] = f2bf(val);
            } else {
                v[(bh * DD + ci) * NN + n] = f2bf(val);
            }
        }
    }
}

// ---------------------------------------------------------------------------
// Kernel 3: flash attention per (b,h). 256 threads = 8 waves; each wave owns
// 16 query rows (block = 128 rows). K (32x64) and V (64x32) chunks double-
// buffered via async global->LDS (one 16B op per thread per buffer), shared
// by all 8 waves and overlapped with compute. Output -> [B,N,C] bf16.
// blocks = (B*H) * 8 = 1024 x 256 threads
// ---------------------------------------------------------------------------
__global__ void la_attn(const __bf16* __restrict__ q,   // [B,h,N,d]
                        const __bf16* __restrict__ k,   // [B,h,N,d]
                        const __bf16* __restrict__ v,   // [B,h,d,N]
                        __bf16* __restrict__ ao) {      // [B,N,C]
    __shared__ __bf16 kt[2][32 * 64];              // [m][c] 4KB x2
    __shared__ __bf16 vt[2][64 * 32];              // [c][m] 4KB x2
    __shared__ __bf16 pstage[8][16 * 32];
    int wid  = threadIdx.x >> 5;
    int lane = threadIdx.x & 31;
    int blk  = blockIdx.x;
    int rb   = blk & 7;
    int bh   = blk >> 3;
    int n0   = rb * 128 + wid * 16;

    const __bf16* qp = q + ((size_t)bh * NN + n0) * DD;
    const __bf16* kp = k + (size_t)bh * NN * DD;
    const __bf16* vp = v + (size_t)bh * DD * NN;

    v16bf aq0 = load_a(qp, DD, lane);        // d = 0..31
    v16bf aq1 = load_a(qp + 32, DD, lane);   // d = 32..63

    v8f o0 = {}, o1 = {}, o2 = {}, o3 = {};
    float mrow[8], lrow[8];
#pragma unroll
    for (int r = 0; r < 8; ++r) { mrow[r] = -1e30f; lrow[r] = 0.f; }

    __bf16* pp = pstage[wid];
    int t   = threadIdx.x;                   // 0..255
    int km  = t >> 3, kpart = t & 7;         // K: 8 x 16B per 64-elem row
    int vr  = t >> 2, vpart = t & 3;         // V: 4 x 16B per 32-elem row
    const __bf16* kg = kp + (size_t)km * DD + kpart * 8;   // + m0*DD
    const __bf16* vg = vp + (size_t)vr * NN + vpart * 8;   // + m0
    int col = lane & 15;
    int hi  = lane >> 4;

    // prologue: stage chunk 0 into buffer 0
    async_ld_b128(lds_off32(kt[0]) + (unsigned)t * 16, kg);
    async_ld_b128(lds_off32(vt[0]) + (unsigned)t * 16, vg);

    for (int m0 = 0, it = 0; m0 < NN; m0 += 32, ++it) {
        wait_async0();
        __syncthreads();
        if (m0 + 32 < NN) {                  // overlap next K/V DMA with compute
            int nb = (it + 1) & 1;
            async_ld_b128(lds_off32(kt[nb]) + (unsigned)t * 16,
                          kg + (size_t)(m0 + 32) * DD);
            async_ld_b128(lds_off32(vt[nb]) + (unsigned)t * 16,
                          vg + (m0 + 32));
        }
        const __bf16* kc = kt[it & 1];
        const __bf16* vc = vt[it & 1];

        // ---- S = Q K^T : element (c, m) at kc[m*64 + c] ----
        v8f s0 = {}, s1 = {};
        s0 = wmma_bf16(aq0, load_b(kc, DD, lane), s0);
        s0 = wmma_bf16(aq1, load_b(kc + 32, DD, lane), s0);
        s1 = wmma_bf16(aq0, load_b(kc + 16 * DD, DD, lane), s1);
        s1 = wmma_bf16(aq1, load_b(kc + 16 * DD + 32, DD, lane), s1);

        // ---- online softmax (row reductions over 16-lane column groups) ----
#pragma unroll
        for (int r = 0; r < 8; ++r) {
            float mx = fmaxf(s0[r], s1[r]);
            mx = fmaxf(mx, __shfl_xor(mx, 1));
            mx = fmaxf(mx, __shfl_xor(mx, 2));
            mx = fmaxf(mx, __shfl_xor(mx, 4));
            mx = fmaxf(mx, __shfl_xor(mx, 8));
            float mnew  = fmaxf(mrow[r], mx);
            float alpha = __expf(mrow[r] - mnew);
            mrow[r] = mnew;
            float p0 = __expf(s0[r] - mnew);
            float p1 = __expf(s1[r] - mnew);
            float ps = p0 + p1;
            ps += __shfl_xor(ps, 1);
            ps += __shfl_xor(ps, 2);
            ps += __shfl_xor(ps, 4);
            ps += __shfl_xor(ps, 8);
            lrow[r] = lrow[r] * alpha + ps;
            o0[r] *= alpha; o1[r] *= alpha; o2[r] *= alpha; o3[r] *= alpha;
            s0[r] = p0; s1[r] = p1;
        }

        // ---- stage P (bf16) to LDS row-major [16][32]: D- -> A-layout ----
#pragma unroll
        for (int r = 0; r < 8; ++r) {
            int row = r + 8 * hi;
            pp[row * 32 + col]      = f2bf(s0[r]);
            pp[row * 32 + 16 + col] = f2bf(s1[r]);
        }
        v16bf apv = load_a(pp, 32, lane);

        // ---- O += P V^T : element (m, c) at vc[c*32 + m] ----
        o0 = wmma_bf16(apv, load_b(vc + 0 * 16 * 32, 32, lane), o0);
        o1 = wmma_bf16(apv, load_b(vc + 1 * 16 * 32, 32, lane), o1);
        o2 = wmma_bf16(apv, load_b(vc + 2 * 16 * 32, 32, lane), o2);
        o3 = wmma_bf16(apv, load_b(vc + 3 * 16 * 32, 32, lane), o3);
    }

    // ---- normalize & store: attn-out [B,N,C], c block = h*64 ----
    int h = bh & 7;
    int b = bh >> 3;
    __bf16* aop = ao + ((size_t)b * NN + n0) * CC + h * DD;
#pragma unroll
    for (int r = 0; r < 8; ++r) {
        int   row  = r + 8 * hi;
        float invl = 1.f / lrow[r];
        __bf16* rp = aop + (size_t)row * CC;
        rp[col]      = f2bf(o0[r] * invl);
        rp[16 + col] = f2bf(o1[r] * invl);
        rp[32 + col] = f2bf(o2[r] * invl);
        rp[48 + col] = f2bf(o3[r] * invl);
    }
}

// ---------------------------------------------------------------------------
// Kernel 4: proj + bias + residual, double-buffered async A staging.
// Block = 8 waves sharing one (b, n-tile), each wave a 64-wide o-strip.
// blocks = 64(nt) * 16(b) = 1024 x 256 threads
// ---------------------------------------------------------------------------
__global__ void la_proj(const __bf16* __restrict__ ao,   // [B,N,C]
                        const __bf16* __restrict__ pw,   // [C,C] bf16 (row o)
                        const float*  __restrict__ pb,   // [C]
                        const float*  __restrict__ x,    // residual [B,C,N]
                        float* __restrict__ out) {       // [B,C,N]
    __shared__ __bf16 astage[2][16 * 32];          // [n][c] 1KB x2
    int blk = blockIdx.x;
    int nt  = blk & 63;
    int b   = blk >> 6;
    int wid  = threadIdx.x >> 5;
    int lane = threadIdx.x & 31;
    int ot4  = wid;                                // 0..7 -> 64-wide o strip

    const __bf16* A  = ao + ((size_t)b * NN + nt * 16) * CC;
    const __bf16* Bb = pw + (size_t)ot4 * 64 * CC;

    int t  = threadIdx.x;
    int sr = t >> 2;                               // 0..15 (t < 64)
    int sp = t & 3;
    const __bf16* gsrc = A + (size_t)sr * CC + sp * 8;

    if (t < 64)
        async_ld_b128(lds_off32(astage[0]) + (unsigned)t * 16, gsrc);

    v8f acc0 = {}, acc1 = {}, acc2 = {}, acc3 = {};
    for (int k0 = 0, it = 0; k0 < CC; k0 += 32, ++it) {
        wait_async0();
        __syncthreads();
        if (t < 64 && k0 + 32 < CC)
            async_ld_b128(lds_off32(astage[(it + 1) & 1]) + (unsigned)t * 16,
                          gsrc + k0 + 32);

        v16bf a = load_a(astage[it & 1], 32, lane);
        acc0 = wmma_bf16(a, load_b(Bb + (size_t) 0 * CC + k0, CC, lane), acc0);
        acc1 = wmma_bf16(a, load_b(Bb + (size_t)16 * CC + k0, CC, lane), acc1);
        acc2 = wmma_bf16(a, load_b(Bb + (size_t)32 * CC + k0, CC, lane), acc2);
        acc3 = wmma_bf16(a, load_b(Bb + (size_t)48 * CC + k0, CC, lane), acc3);
    }

    int col = lane & 15;
    int hi  = lane >> 4;
    v8f* accs[4] = { &acc0, &acc1, &acc2, &acc3 };
#pragma unroll
    for (int j = 0; j < 4; ++j) {
        int   o    = ot4 * 64 + j * 16 + col;
        float bias = pb[o];
        const float* xb = x   + ((size_t)b * CC + o) * NN + nt * 16 + 8 * hi;
        float*       ob = out + ((size_t)b * CC + o) * NN + nt * 16 + 8 * hi;
        v8f av = *accs[j];
        float4 w0, w1;
        w0.x = av[0] + bias + xb[0];
        w0.y = av[1] + bias + xb[1];
        w0.z = av[2] + bias + xb[2];
        w0.w = av[3] + bias + xb[3];
        w1.x = av[4] + bias + xb[4];
        w1.y = av[5] + bias + xb[5];
        w1.z = av[6] + bias + xb[6];
        w1.w = av[7] + bias + xb[7];
        *(float4*)(ob)     = w0;
        *(float4*)(ob + 4) = w1;
    }
}

// ---------------------------------------------------------------------------
// Launch
// ---------------------------------------------------------------------------
extern "C" void kernel_launch(void* const* d_in, const int* in_sizes, int n_in,
                              void* d_out, int out_size, void* d_ws, size_t ws_size,
                              hipStream_t stream) {
    const float* x      = (const float*)d_in[0];
    const float* gn_w   = (const float*)d_in[1];
    const float* gn_b   = (const float*)d_in[2];
    const float* qkv_w  = (const float*)d_in[3];
    const float* qkv_b  = (const float*)d_in[4];
    const float* proj_w = (const float*)d_in[5];
    const float* proj_b = (const float*)d_in[6];
    float* out = (float*)d_out;

    const size_t TEN = (size_t)BB * CC * NN;      // 8,388,608 elements
    __bf16* ws  = (__bf16*)d_ws;
    __bf16* xn  = ws;                             // [B,N,C]
    __bf16* qb  = xn + TEN;                       // [B,h,N,d]
    __bf16* kb  = qb + TEN;                       // [B,h,N,d]
    __bf16* vb  = kb + TEN;                       // [B,h,d,N]
    __bf16* aob = vb + TEN;                       // [B,N,C]
    __bf16* wqb = aob + TEN;                      // [1536,512]
    __bf16* wpb = wqb + (size_t)O3 * CC;          // [512,512]

    la_cvt_bf16<<<(O3 * CC + 255) / 256, 256, 0, stream>>>(qkv_w, wqb, O3 * CC);
    la_cvt_bf16<<<(CC * CC + 255) / 256, 256, 0, stream>>>(proj_w, wpb, CC * CC);

    la_groupnorm<<<BB * HH, 256, 0, stream>>>(x, gn_w, gn_b, xn);

    la_qkv<<<3072, 256, 0, stream>>>(wqb, qkv_b, xn, qb, kb, vb);

    la_attn<<<1024, 256, 0, stream>>>(qb, kb, vb, aob);

    la_proj<<<1024, 256, 0, stream>>>(aob, wpb, proj_b, x, out);
}
